// Attentive_context_sr_42021960024113
// MI455X (gfx1250) — matrix operand order repaired and checked by reading, compiled, not run on hardware
//
#include <hip/hip_runtime.h>

#define DEVFN __device__ __forceinline__

typedef __attribute__((ext_vector_type(16))) _Float16 v16h;
typedef __attribute__((ext_vector_type(8)))  float    v8f;

union F16Frag { v16h v; uint4 q[2]; };

constexpr int S_ = 512, B_ = 32, IN_ = 512, H_ = 256;
constexpr int H2_ = 512, H3_ = 768, H4_ = 1024;
constexpr int NSEC_ = 16, SENT_ = 100, DOC_ = 200;

DEVFN float sigmoidf_(float x) { return 1.0f / (1.0f + __expf(-x)); }

DEVFN float dotn(const float* a, const float* b, int n) {
    float s = 0.0f;
    for (int i = 0; i < n; ++i) s += a[i] * b[i];
    return s;
}

// A fragment (16x32 f16, M x K): lane l -> row m = l%16.
// holds contiguous K = 16*(l/16) + {0..15}.
DEVFN v16h load_frag_a(const _Float16* base, int ld, int mbase, int kb, int lane) {
    F16Frag f;
    const _Float16* p = base + (size_t)(mbase + (lane & 15)) * ld + kb + ((lane >> 4) << 4);
    f.q[0] = *reinterpret_cast<const uint4*>(p);
    f.q[1] = *reinterpret_cast<const uint4*>(p + 8);
    return f.v;
}

// B fragment (32x16 f16, K x N) from row-major W[N][K] (K contiguous):
// lane l -> col n = l%16, holds contiguous K = 16*(l/16) + {0..15}.
DEVFN v16h load_frag_b_fast(const _Float16* base, int ld, int nbase, int kb, int lane) {
    F16Frag f;
    const _Float16* p = base + (size_t)(nbase + (lane & 15)) * ld + kb + ((lane >> 4) << 4);
    f.q[0] = *reinterpret_cast<const uint4*>(p);
    f.q[1] = *reinterpret_cast<const uint4*>(p + 8);
    return f.v;
}

// Per-lane guarded variant (only instantiated for the N=100 GEMM).
DEVFN v16h load_frag_b_guard(const _Float16* base, int ld, int nbase, int kb, int lane, int nmax) {
    int n = nbase + (lane & 15);
    F16Frag f;
    if (n < nmax) {
        const _Float16* p = base + (size_t)n * ld + kb + ((lane >> 4) << 4);
        f.q[0] = *reinterpret_cast<const uint4*>(p);
        f.q[1] = *reinterpret_cast<const uint4*>(p + 8);
    } else {
        f.q[0] = make_uint4(0u, 0u, 0u, 0u);
        f.q[1] = make_uint4(0u, 0u, 0u, 0u);
    }
    return f.v;
}

DEVFN v8f wmma_f16(v16h a, v16h b, v8f c) {
    return __builtin_amdgcn_wmma_f32_16x16x32_f16(false, a, false, b, (short)0, c, false, false);
}

// ---------------------------------------------------------------------------
// Utility kernels
// ---------------------------------------------------------------------------
__global__ void __launch_bounds__(256)
k_cvt_f16(const float* __restrict__ src, _Float16* __restrict__ dst, int n) {
    int i = blockIdx.x * 256 + threadIdx.x;
    if (i < n) dst[i] = (_Float16)src[i];
}

__global__ void __launch_bounds__(256)
k_zero_f32(float* __restrict__ p, int n) {
    int i = blockIdx.x * 256 + threadIdx.x;
    if (i < n) p[i] = 0.0f;
}

// ---------------------------------------------------------------------------
// Generic WMMA GEMM: C[M,N] = A[M,K] * W[N,K]^T  (A,W f16; C f32)
// MODE 0: C = acc + bias       MODE 1: C = relu(acc + bias)
// MODE 2: scoreOut[row] += sum_n acc[row,n] * cv[n]   (fused context-vector dot)
// GUARD: compile-time; false -> branch-free inner loop (N multiple of 64).
// Each wave computes a 16x64 strip (4 accumulators).
// ---------------------------------------------------------------------------
template <int MODE, bool GUARD>
__global__ void __launch_bounds__(256)
k_wmma_gemm(const _Float16* __restrict__ A, const _Float16* __restrict__ W,
            const float* __restrict__ bias, float* __restrict__ C,
            float* __restrict__ scoreOut, const float* __restrict__ cv,
            int M, int N, int K) {
    const int lane = threadIdx.x & 31;
    const int wave = threadIdx.x >> 5;
    const int nt4  = (N + 63) >> 6;
    const int tile = blockIdx.x * 8 + wave;
    const int mt   = tile / nt4;
    const int nt   = tile - mt * nt4;
    if (mt * 16 >= M) return;
    const int mbase = mt * 16;

    v8f acc[4] = {};
    for (int kb = 0; kb < K; kb += 32) {
        v16h a = load_frag_a(A, K, mbase, kb, lane);
        // speculative prefetch of the next A fragment (safe past end of row)
        __builtin_prefetch(A + (size_t)(mbase + (lane & 15)) * K + kb + 32, 0, 1);
#pragma unroll
        for (int j = 0; j < 4; ++j) {
            v16h b;
            if (GUARD)
                b = load_frag_b_guard(W, K, nt * 64 + j * 16, kb, lane, N);
            else
                b = load_frag_b_fast(W, K, nt * 64 + j * 16, kb, lane);
            acc[j] = wmma_f16(a, b, acc[j]);
        }
    }

    // C/D layout: lane l holds column n = l%16, rows m = 8*(l/16) + r (r = VGPR idx)
    const int nlo  = lane & 15;
    const int mrow = mbase + ((lane >> 4) << 3);
#pragma unroll
    for (int j = 0; j < 4; ++j) {
        int n = nt * 64 + j * 16 + nlo;
        if (GUARD && n >= N) continue;
        if (MODE == 2) {
            float cvv = cv[n];
#pragma unroll
            for (int r = 0; r < 8; ++r)
                atomicAdd(scoreOut + (mrow + r), acc[j][r] * cvv);
        } else {
            float bb = bias[n];
#pragma unroll
            for (int r = 0; r < 8; ++r) {
                float v = acc[j][r] + bb;
                if (MODE == 1) v = fmaxf(v, 0.0f);
                C[(size_t)(mrow + r) * N + n] = v;
            }
        }
    }
}

// ---------------------------------------------------------------------------
// Persistent bidirectional GRU recurrence. blockIdx.x = direction (0 fwd, 1 bwd).
// h kept in LDS as f16 (double buffered) for WMMA A operand; each wave owns
// two 16-column groups and computes gh = h @ Whh^T for its r/z/n tiles with
// WMMA, then does the gate math and updates h.  bwd direction reads gi
// reversed and stores hs already re-reversed (hs_b[s] = state fed by input s).
// ---------------------------------------------------------------------------
__global__ void __launch_bounds__(256)
k_gru_scan(const float* __restrict__ gi_f, const float* __restrict__ gi_b,
           const _Float16* __restrict__ Whh_f16, const _Float16* __restrict__ Whh_b16,
           const float* __restrict__ bhh_f, const float* __restrict__ bhh_b,
           float* __restrict__ hs_f, float* __restrict__ hs_b) {
    __shared__ __align__(32) _Float16 h16[2][B_ * H_];

    const int dir = blockIdx.x;
    const float* gi      = dir ? gi_b : gi_f;
    const _Float16* Whh  = dir ? Whh_b16 : Whh_f16;
    const float* bhh     = dir ? bhh_b : bhh_f;
    float* hs            = dir ? hs_b : hs_f;

    const int tid  = threadIdx.x;
    const int lane = tid & 31;
    const int wave = tid >> 5;

    for (int i = tid; i < B_ * H_; i += 256) h16[0][i] = (_Float16)0.0f;

    float hold[2][2][8];
#pragma unroll
    for (int g = 0; g < 2; ++g)
#pragma unroll
        for (int m = 0; m < 2; ++m)
#pragma unroll
            for (int r = 0; r < 8; ++r) hold[g][m][r] = 0.0f;

    __syncthreads();

    for (int t = 0; t < S_; ++t) {
        const int s = dir ? (S_ - 1 - t) : t;
        const float* gis = gi + (size_t)s * (B_ * H3_);
        const _Float16* hrd = h16[t & 1];
        _Float16* hwr = h16[(t & 1) ^ 1];

#pragma unroll
        for (int g2 = 0; g2 < 2; ++g2) {
            const int j = (wave * 2 + g2) * 16;
            v8f accR[2] = {}, accZ[2] = {}, accN[2] = {};
            for (int kb = 0; kb < H_; kb += 32) {
                v16h a0 = load_frag_a(hrd, H_, 0,  kb, lane);
                v16h a1 = load_frag_a(hrd, H_, 16, kb, lane);
                v16h br = load_frag_b_fast(Whh, H_, j,          kb, lane);
                v16h bz = load_frag_b_fast(Whh, H_, H_ + j,     kb, lane);
                v16h bn = load_frag_b_fast(Whh, H_, 2 * H_ + j, kb, lane);
                accR[0] = wmma_f16(a0, br, accR[0]);
                accR[1] = wmma_f16(a1, br, accR[1]);
                accZ[0] = wmma_f16(a0, bz, accZ[0]);
                accZ[1] = wmma_f16(a1, bz, accZ[1]);
                accN[0] = wmma_f16(a0, bn, accN[0]);
                accN[1] = wmma_f16(a1, bn, accN[1]);
            }
            const int col = j + (lane & 15);
            const int mo  = (lane >> 4) << 3;
            const float bhr = bhh[col];
            const float bhz = bhh[H_ + col];
            const float bhn = bhh[2 * H_ + col];
#pragma unroll
            for (int mt = 0; mt < 2; ++mt) {
#pragma unroll
                for (int r = 0; r < 8; ++r) {
                    const int b = mt * 16 + mo + r;
                    const float* gb = gis + (size_t)b * H3_;
                    float rr = sigmoidf_(gb[col] + accR[mt][r] + bhr);
                    float zz = sigmoidf_(gb[H_ + col] + accZ[mt][r] + bhz);
                    float nn = tanhf(gb[2 * H_ + col] + rr * (accN[mt][r] + bhn));
                    float hn = (1.0f - zz) * nn + zz * hold[g2][mt][r];
                    hold[g2][mt][r] = hn;
                    hwr[b * H_ + col] = (_Float16)hn;
                    hs[((size_t)s * B_ + b) * H_ + col] = hn;
                }
            }
        }
        __syncthreads();
    }
}

// ---------------------------------------------------------------------------
// lc[b,n,0:2H]: begin/end gather-diff on 1-padded RNN outputs
// ---------------------------------------------------------------------------
__global__ void __launch_bounds__(256)
k_lc(const float* __restrict__ hsF, const float* __restrict__ hsB,
     const int* __restrict__ begin, const int* __restrict__ endp,
     float* __restrict__ lc) {
    int idx = blockIdx.x * 256 + threadIdx.x;
    if (idx >= B_ * NSEC_ * H2_) return;
    int h = idx & (H2_ - 1);
    int n = (idx >> 9) & (NSEC_ - 1);
    int b = idx >> 13;
    int bg = begin[b * NSEC_ + n];
    int en = endp[b * NSEC_ + n];
    float v;
    if (h < H_) {
        float fe = (en >= 1) ? hsF[((size_t)(en - 1) * B_ + b) * H_ + h] : 0.0f;
        float fb = (bg >= 1) ? hsF[((size_t)(bg - 1) * B_ + b) * H_ + h] : 0.0f;
        v = fe - fb;
    } else {
        int hh = h - H_;
        float vb = hsB[((size_t)bg * B_ + b) * H_ + hh];
        float ve = hsB[((size_t)en * B_ + b) * H_ + hh];
        v = vb - ve;
    }
    lc[idx] = v;
}

// ---------------------------------------------------------------------------
// local + tanh-packed attention input rows (f16):
// row (s*B+b)*2+k of [4H]: [context_k | output], k=0 local, k=1 doc
// ---------------------------------------------------------------------------
__global__ void __launch_bounds__(256)
k_build_attn(const float* __restrict__ si, const float* __restrict__ lc,
             const float* __restrict__ hsF, const float* __restrict__ hsB,
             float* __restrict__ localv, _Float16* __restrict__ attn16) {
    int idx = blockIdx.x * 256 + threadIdx.x;
    if (idx >= S_ * B_ * H2_) return;
    int h = idx & (H2_ - 1);
    int b = (idx >> 9) & (B_ - 1);
    int s = idx >> 14;

    float loc = 0.0f;
    const float* sib = si + ((size_t)b * S_ + s) * NSEC_;
    const float* lcb = lc + ((size_t)b * NSEC_) * H2_ + h;
#pragma unroll
    for (int n = 0; n < NSEC_; ++n) loc += sib[n] * lcb[(size_t)n * H2_];
    localv[idx] = loc;

    float dv = (h < H_) ? hsF[((size_t)(S_ - 1) * B_ + b) * H_ + h]
                        : hsB[(size_t)b * H_ + (h - H_)];
    float ov = (h < H_) ? hsF[((size_t)s * B_ + b) * H_ + h]
                        : hsB[((size_t)s * B_ + b) * H_ + (h - H_)];

    size_t row0 = ((size_t)(s * B_ + b) * 2) * H4_;
    attn16[row0 + h]        = (_Float16)tanhf(loc);
    attn16[row0 + H4_ + h]  = (_Float16)tanhf(dv);
    _Float16 to = (_Float16)tanhf(ov);
    attn16[row0 + H2_ + h]       = to;
    attn16[row0 + H4_ + H2_ + h] = to;
}

// ---------------------------------------------------------------------------
// softmax over k=2, weighted context, pack mlp_in rows (f16): [weighted|output]
// ---------------------------------------------------------------------------
__global__ void __launch_bounds__(256)
k_build_mlp(const float* __restrict__ scores, const float* __restrict__ localv,
            const float* __restrict__ hsF, const float* __restrict__ hsB,
            _Float16* __restrict__ mlp16) {
    int idx = blockIdx.x * 256 + threadIdx.x;
    if (idx >= S_ * B_ * H2_) return;
    int h = idx & (H2_ - 1);
    int b = (idx >> 9) & (B_ - 1);
    int s = idx >> 14;
    int row = s * B_ + b;

    float s0 = scores[row * 2], s1 = scores[row * 2 + 1];
    float mx = fmaxf(s0, s1);
    float e0 = __expf(s0 - mx), e1 = __expf(s1 - mx);
    float inv = 1.0f / (e0 + e1);
    float w0 = e0 * inv, w1 = e1 * inv;

    float dv = (h < H_) ? hsF[((size_t)(S_ - 1) * B_ + b) * H_ + h]
                        : hsB[(size_t)b * H_ + (h - H_)];
    float wv = w0 * localv[idx] + w1 * dv;
    float ov = (h < H_) ? hsF[((size_t)s * B_ + b) * H_ + h]
                        : hsB[((size_t)s * B_ + b) * H_ + (h - H_)];

    mlp16[(size_t)row * H4_ + h]       = (_Float16)wv;
    mlp16[(size_t)row * H4_ + H2_ + h] = (_Float16)ov;
}

// ---------------------------------------------------------------------------
// small dense epilogues
// ---------------------------------------------------------------------------
__global__ void __launch_bounds__(256)
k_content(const float* __restrict__ ss, const float* __restrict__ cw,
          const float* __restrict__ cb, float* __restrict__ content) {
    int idx = blockIdx.x * 256 + threadIdx.x;
    if (idx >= S_ * B_) return;
    int b = idx & (B_ - 1), s = idx >> 5;
    content[b * S_ + s] = dotn(ss + (size_t)idx * SENT_, cw, SENT_) + cb[0];
}

__global__ void __launch_bounds__(256)
k_avg(const float* __restrict__ ss, const int* __restrict__ len,
      float* __restrict__ avg) {
    int idx = blockIdx.x * 256 + threadIdx.x;
    if (idx >= B_ * SENT_) return;
    int b = idx / SENT_, k = idx - b * SENT_;
    float a = 0.0f;
    for (int s = 0; s < S_; ++s) a += ss[((size_t)s * B_ + b) * SENT_ + k];
    avg[idx] = a / (float)len[b];
}

__global__ void __launch_bounds__(256)
k_doc1(const float* __restrict__ avg, const float* __restrict__ W1,
       const float* __restrict__ b1, float* __restrict__ tmp) {
    int idx = blockIdx.x * 256 + threadIdx.x;
    if (idx >= B_ * DOC_) return;
    int b = idx / DOC_, j = idx - b * DOC_;
    tmp[idx] = tanhf(dotn(avg + b * SENT_, W1 + (size_t)j * SENT_, SENT_) + b1[j]);
}

__global__ void __launch_bounds__(256)
k_doc2(const float* __restrict__ tmp, const float* __restrict__ W2,
       const float* __restrict__ b2, float* __restrict__ dvec) {
    int idx = blockIdx.x * 256 + threadIdx.x;
    if (idx >= B_ * SENT_) return;
    int b = idx / SENT_, k = idx - b * SENT_;
    dvec[idx] = dotn(tmp + b * DOC_, W2 + (size_t)k * DOC_, DOC_) + b2[k];
}

__global__ void __launch_bounds__(256)
k_static(const float* __restrict__ ss, const float* __restrict__ dvec,
         const float* __restrict__ content, const int* __restrict__ len,
         const float* __restrict__ pos_emb, const float* __restrict__ pos_W,
         const float* __restrict__ seg_emb, const float* __restrict__ seg_W,
         const float* __restrict__ biasp, float* __restrict__ staticv) {
    int idx = blockIdx.x * 256 + threadIdx.x;
    if (idx >= S_ * B_) return;
    int b = idx & (B_ - 1), s = idx >> 5;

    float sal = dotn(ss + (size_t)idx * SENT_, dvec + b * SENT_, SENT_);
    int ap = s + 1;
    int absc = ap > 25 ? 25 : ap;
    float posl = dotn(pos_emb + absc * 50, pos_W, 50);
    float chunk = rintf((float)len[b] * 0.25f);
    int rel = (int)ceilf((float)ap / chunk);
    rel = rel < 0 ? 0 : (rel > 4 ? 4 : rel);
    float segl = dotn(seg_emb + rel * 50, seg_W, 50);
    staticv[b * S_ + s] = content[b * S_ + s] + sal + posl + segl + biasp[0];
}

// ---------------------------------------------------------------------------
// final sequential novelty scan (single workgroup, LDS-resident summary)
// ---------------------------------------------------------------------------
__global__ void __launch_bounds__(256)
k_final_scan(const float* __restrict__ ss, const float* __restrict__ staticv,
             const float* __restrict__ simW, float* __restrict__ out) {
    __shared__ float summary[B_ * SENT_];
    __shared__ float tsum[B_ * SENT_];
    __shared__ float vv[B_ * SENT_];
    __shared__ float lgsh[B_];
    int tid = threadIdx.x;
    for (int i = tid; i < B_ * SENT_; i += 256) summary[i] = 0.0f;
    __syncthreads();

    for (int s = 0; s < S_; ++s) {
        const float* st = ss + (size_t)s * B_ * SENT_;
        for (int i = tid; i < B_ * SENT_; i += 256) tsum[i] = tanhf(summary[i]);
        __syncthreads();
        for (int i = tid; i < B_ * SENT_; i += 256) {
            int b = i / SENT_, ii = i - b * SENT_;
            const float* wr = simW + (size_t)ii * SENT_;
            const float* tb = tsum + b * SENT_;
            float acc = 0.0f;
            for (int j = 0; j < SENT_; ++j) acc += wr[j] * tb[j];
            vv[i] = acc;
        }
        __syncthreads();
        if (tid < B_) {
            int b = tid;
            const float* sb = st + b * SENT_;
            const float* vb = vv + b * SENT_;
            float nov = 0.0f;
            for (int i = 0; i < SENT_; ++i) nov += sb[i] * vb[i];
            float lg = staticv[b * S_ + s] - nov;
            lgsh[b] = lg;
            out[(size_t)s * B_ + b] = lg;
        }
        __syncthreads();
        for (int i = tid; i < B_ * SENT_; i += 256) {
            int b = i / SENT_;
            summary[i] += st[i] * sigmoidf_(lgsh[b]);
        }
        __syncthreads();
    }
}

// ---------------------------------------------------------------------------
extern "C" void kernel_launch(void* const* d_in, const int* in_sizes, int n_in,
                              void* d_out, int out_size, void* d_ws, size_t ws_size,
                              hipStream_t stream) {
    (void)in_sizes; (void)n_in; (void)out_size; (void)ws_size;

    const float* inputs      = (const float*)d_in[0];
    const int*   input_len   = (const int*)d_in[1];
    const float* section_ind = (const float*)d_in[2];
    const int*   begin       = (const int*)d_in[3];
    const int*   endp        = (const int*)d_in[4];
    const float* Wih_f = (const float*)d_in[5];
    const float* Whh_f = (const float*)d_in[6];
    const float* bih_f = (const float*)d_in[7];
    const float* bhh_f = (const float*)d_in[8];
    const float* Wih_b = (const float*)d_in[9];
    const float* Whh_b = (const float*)d_in[10];
    const float* bih_b = (const float*)d_in[11];
    const float* bhh_b = (const float*)d_in[12];
    const float* feat_attn   = (const float*)d_in[13];
    const float* context_vec = (const float*)d_in[14];
    const float* sent_W = (const float*)d_in[15];
    const float* sent_b = (const float*)d_in[16];
    const float* content_W = (const float*)d_in[17];
    const float* content_b = (const float*)d_in[18];
    const float* doc_W1 = (const float*)d_in[19];
    const float* doc_b1 = (const float*)d_in[20];
    const float* doc_W2 = (const float*)d_in[21];
    const float* doc_b2 = (const float*)d_in[22];
    const float* sim_W  = (const float*)d_in[23];
    const float* biasp  = (const float*)d_in[24];
    const float* pos_emb = (const float*)d_in[25];
    const float* pos_W   = (const float*)d_in[26];
    const float* seg_emb = (const float*)d_in[27];
    const float* seg_W   = (const float*)d_in[28];
    float* out = (float*)d_out;

    char* base = (char*)d_ws;
    size_t off = 0;
    auto alloc = [&](size_t bytes) -> char* {
        char* p = base + off;
        off = (off + bytes + 255) & ~(size_t)255;
        return p;
    };
    _Float16* X16    = (_Float16*)alloc((size_t)S_ * B_ * IN_ * 2);
    _Float16* WihF16 = (_Float16*)alloc((size_t)H3_ * IN_ * 2);
    _Float16* WihB16 = (_Float16*)alloc((size_t)H3_ * IN_ * 2);
    _Float16* WhhF16 = (_Float16*)alloc((size_t)H3_ * H_ * 2);
    _Float16* WhhB16 = (_Float16*)alloc((size_t)H3_ * H_ * 2);
    _Float16* Feat16 = (_Float16*)alloc((size_t)H4_ * H4_ * 2);
    _Float16* Sent16 = (_Float16*)alloc((size_t)SENT_ * H4_ * 2);
    float* giF    = (float*)alloc((size_t)S_ * B_ * H3_ * 4);
    float* giB    = (float*)alloc((size_t)S_ * B_ * H3_ * 4);
    float* hsF    = (float*)alloc((size_t)S_ * B_ * H_ * 4);
    float* hsB    = (float*)alloc((size_t)S_ * B_ * H_ * 4);
    float* lc     = (float*)alloc((size_t)B_ * NSEC_ * H2_ * 4);
    float* localv = (float*)alloc((size_t)S_ * B_ * H2_ * 4);
    _Float16* Attn16 = (_Float16*)alloc((size_t)S_ * B_ * 2 * H4_ * 2);
    float* scores = (float*)alloc((size_t)S_ * B_ * 2 * 4);
    _Float16* Mlp16  = (_Float16*)alloc((size_t)S_ * B_ * H4_ * 2);
    float* ss      = (float*)alloc((size_t)S_ * B_ * SENT_ * 4);
    float* content = (float*)alloc((size_t)B_ * S_ * 4);
    float* avg     = (float*)alloc((size_t)B_ * SENT_ * 4);
    float* tmpdoc  = (float*)alloc((size_t)B_ * DOC_ * 4);
    float* dvec    = (float*)alloc((size_t)B_ * SENT_ * 4);
    float* staticv = (float*)alloc((size_t)B_ * S_ * 4);

    auto CVT = [&](const float* s, _Float16* d, int n) {
        k_cvt_f16<<<(n + 255) / 256, 256, 0, stream>>>(s, d, n);
    };
    CVT(inputs, X16, S_ * B_ * IN_);
    CVT(Wih_f, WihF16, H3_ * IN_);
    CVT(Wih_b, WihB16, H3_ * IN_);
    CVT(Whh_f, WhhF16, H3_ * H_);
    CVT(Whh_b, WhhB16, H3_ * H_);
    CVT(feat_attn, Feat16, H4_ * H4_);
    CVT(sent_W, Sent16, SENT_ * H4_);

    auto gemm_blocks = [](int M, int N) {
        int nt4 = (N + 63) >> 6;
        int tiles = (M / 16) * nt4;
        return (tiles + 7) / 8;
    };

    // GRU input projections (N=768, branch-free instantiation)
    k_wmma_gemm<0, false><<<gemm_blocks(S_ * B_, H3_), 256, 0, stream>>>(
        X16, WihF16, bih_f, giF, nullptr, nullptr, S_ * B_, H3_, IN_);
    k_wmma_gemm<0, false><<<gemm_blocks(S_ * B_, H3_), 256, 0, stream>>>(
        X16, WihB16, bih_b, giB, nullptr, nullptr, S_ * B_, H3_, IN_);

    // bidirectional recurrence (persistent, WMMA per step)
    k_gru_scan<<<2, 256, 0, stream>>>(giF, giB, WhhF16, WhhB16, bhh_f, bhh_b, hsF, hsB);

    // section gather-diff
    k_lc<<<(B_ * NSEC_ * H2_ + 255) / 256, 256, 0, stream>>>(hsF, hsB, begin, endp, lc);

    // attention input build
    k_build_attn<<<(S_ * B_ * H2_ + 255) / 256, 256, 0, stream>>>(
        section_ind, lc, hsF, hsB, localv, Attn16);

    // attention scores: proj = attn_in @ feat_attn^T, fused . context_vector
    k_zero_f32<<<(S_ * B_ * 2 + 255) / 256, 256, 0, stream>>>(scores, S_ * B_ * 2);
    k_wmma_gemm<2, false><<<gemm_blocks(S_ * B_ * 2, H4_), 256, 0, stream>>>(
        Attn16, Feat16, nullptr, nullptr, scores, context_vec, S_ * B_ * 2, H4_, H4_);

    // softmax + weighted context + mlp input
    k_build_mlp<<<(S_ * B_ * H2_ + 255) / 256, 256, 0, stream>>>(
        scores, localv, hsF, hsB, Mlp16);

    // ss = relu(mlp_in @ sent_W^T + sent_b)   (N=100 -> guarded instantiation)
    k_wmma_gemm<1, true><<<gemm_blocks(S_ * B_, SENT_), 256, 0, stream>>>(
        Mlp16, Sent16, sent_b, ss, nullptr, nullptr, S_ * B_, SENT_, H4_);

    // epilogues
    k_content<<<(S_ * B_ + 255) / 256, 256, 0, stream>>>(ss, content_W, content_b, content);
    k_avg<<<(B_ * SENT_ + 255) / 256, 256, 0, stream>>>(ss, input_len, avg);
    k_doc1<<<(B_ * DOC_ + 255) / 256, 256, 0, stream>>>(avg, doc_W1, doc_b1, tmpdoc);
    k_doc2<<<(B_ * SENT_ + 255) / 256, 256, 0, stream>>>(tmpdoc, doc_W2, doc_b2, dvec);
    k_static<<<(S_ * B_ + 255) / 256, 256, 0, stream>>>(
        ss, dvec, content, input_len, pos_emb, pos_W, seg_emb, seg_W, biasp, staticv);

    // final selection scan
    k_final_scan<<<1, 256, 0, stream>>>(ss, staticv, sim_W, out);
}
